// TimeMIL_SingleSensor_v1_670014898675
// MI455X (gfx1250) — matrix operand order repaired and checked by reading, not compiled, not run on hardware
//
#include <hip/hip_runtime.h>
#include <math.h>

typedef __attribute__((ext_vector_type(16))) _Float16 v16h;
typedef __attribute__((ext_vector_type(8)))  _Float16 v8h;
typedef __attribute__((ext_vector_type(8)))  float    v8f;

static inline unsigned divup(long a, long b) { return (unsigned)((a + b - 1) / b); }

// ---------------------------------------------------------------------------
// Batched WMMA GEMM:  C[M,N] = alpha * A[M,K] @ B (+bias[n]) (optional relu)
// B is [K,N] (transB=0) or [N,K] (transB=1), row-major, leading dim ldb.
// Batch via blockIdx.z with element strides sA/sB/sC (sA==0 broadcasts A).
// Block tile: 64(M) x 64(N), 4 waves. Each wave owns a 16-wide N slice and
// computes four 16x16 WMMA tiles per K-step, reusing its B fragment.
// LDS: A panel row-major (64x32), B panel N-major (64n x 32k) so fragment
// gathers are contiguous 128-bit ds loads.
// ---------------------------------------------------------------------------
__global__ __launch_bounds__(128)
void wmma_gemm(const float* __restrict__ Ag, const float* __restrict__ Bg,
               float* __restrict__ Cg,
               int M, int N, int K,
               long lda, long ldb, long ldc,
               long sA, long sB, long sC,
               int transB,
               const float* __restrict__ bias,
               float alpha, int relu)
{
    __shared__ _Float16 Ash[64][40];   // [m][k]  row pitch 80B (16B aligned)
    __shared__ _Float16 Bsh[64][40];   // [n][k]  row pitch 80B (16B aligned)

    const int tid  = threadIdx.x;
    const int lane = tid & 31;
    const int wave = tid >> 5;

    const long batch = blockIdx.z;
    const float* A = Ag + batch * sA;
    const float* B = Bg + batch * sB;
    float*       C = Cg + batch * sC;

    const int tileM = blockIdx.y * 64;
    const int tileN = blockIdx.x * 64;

    v8f acc0 = {}, acc1 = {}, acc2 = {}, acc3 = {};

    const int m16 = lane & 15;
    const int kbA = (lane < 16) ? 0 : 8;    // A frag: elems 0..7 -> K=kbA.., 8..15 -> K=kbA+16..
    const int kbB = (lane < 16) ? 0 : 8;    // B frag: elems 0..7 -> K=kbB.., 8..15 -> K=kbB+16..
    const int nrow = (wave << 4) + m16;     // this lane's N column within block

    for (int k0 = 0; k0 < K; k0 += 32) {
        const bool fullK = (k0 + 32 <= K);
        // ---- stage A (64 x 32) ----
        if (fullK && (tileM + 64 <= M)) {
            for (int i = tid; i < 64 * 32; i += 128) {
                int r = i >> 5, c = i & 31;
                Ash[r][c] = (_Float16)A[(long)(tileM + r) * lda + (k0 + c)];
            }
        } else {
            for (int i = tid; i < 64 * 32; i += 128) {
                int r = i >> 5, c = i & 31;
                int gm = tileM + r, gk = k0 + c;
                bool ok = (gm < M) && (gk < K);
                long addr = ok ? ((long)gm * lda + gk) : 0;
                float v = A[addr];
                Ash[r][c] = (_Float16)(ok ? v : 0.f);
            }
        }
        // ---- stage B into Bsh[n][k] ----
        if (!transB) {
            if (fullK && (tileN + 64 <= N)) {
                for (int i = tid; i < 64 * 32; i += 128) {
                    int r = i >> 6, c = i & 63;   // r=k (coalesced over n)
                    Bsh[c][r] = (_Float16)B[(long)(k0 + r) * ldb + (tileN + c)];
                }
            } else {
                for (int i = tid; i < 64 * 32; i += 128) {
                    int r = i >> 6, c = i & 63;
                    int gk = k0 + r, gn = tileN + c;
                    bool ok = (gk < K) && (gn < N);
                    long addr = ok ? ((long)gk * ldb + gn) : 0;
                    float v = B[addr];
                    Bsh[c][r] = (_Float16)(ok ? v : 0.f);
                }
            }
        } else {
            if (fullK && (tileN + 64 <= N)) {
                for (int i = tid; i < 64 * 32; i += 128) {
                    int c = i >> 5, r = i & 31;   // r=k fastest (coalesced over k)
                    Bsh[c][r] = (_Float16)B[(long)(tileN + c) * ldb + (k0 + r)];
                }
            } else {
                for (int i = tid; i < 64 * 32; i += 128) {
                    int c = i >> 5, r = i & 31;
                    int gk = k0 + r, gn = tileN + c;
                    bool ok = (gk < K) && (gn < N);
                    long addr = ok ? ((long)gn * ldb + gk) : 0;
                    float v = B[addr];
                    Bsh[c][r] = (_Float16)(ok ? v : 0.f);
                }
            }
        }
        __syncthreads();

        // prefetch next A K-panel (global_prefetch_b8)
        if (k0 + 32 < K && tid < 64 && (tileM + tid) < M)
            __builtin_prefetch(&A[(long)(tileM + tid) * lda + k0 + 32], 0, 1);

        // B fragment (reused by 4 WMMAs): contiguous 2x128b ds loads
        v8h b0 = *(const v8h*)&Bsh[nrow][kbB];
        v8h b1 = *(const v8h*)&Bsh[nrow][kbB + 16];
        v16h bfrag = __builtin_shufflevector(b0, b1,
            0, 1, 2, 3, 4, 5, 6, 7, 8, 9, 10, 11, 12, 13, 14, 15);

        // 4 M-subtiles, each: 2x128b ds loads + 1 WMMA
        {
            v8h a0 = *(const v8h*)&Ash[m16][kbA];
            v8h a1 = *(const v8h*)&Ash[m16][kbA + 16];
            v16h af = __builtin_shufflevector(a0, a1,
                0, 1, 2, 3, 4, 5, 6, 7, 8, 9, 10, 11, 12, 13, 14, 15);
            acc0 = __builtin_amdgcn_wmma_f32_16x16x32_f16(false, af, false, bfrag, (short)0, acc0, false, false);
        }
        {
            v8h a0 = *(const v8h*)&Ash[16 + m16][kbA];
            v8h a1 = *(const v8h*)&Ash[16 + m16][kbA + 16];
            v16h af = __builtin_shufflevector(a0, a1,
                0, 1, 2, 3, 4, 5, 6, 7, 8, 9, 10, 11, 12, 13, 14, 15);
            acc1 = __builtin_amdgcn_wmma_f32_16x16x32_f16(false, af, false, bfrag, (short)0, acc1, false, false);
        }
        {
            v8h a0 = *(const v8h*)&Ash[32 + m16][kbA];
            v8h a1 = *(const v8h*)&Ash[32 + m16][kbA + 16];
            v16h af = __builtin_shufflevector(a0, a1,
                0, 1, 2, 3, 4, 5, 6, 7, 8, 9, 10, 11, 12, 13, 14, 15);
            acc2 = __builtin_amdgcn_wmma_f32_16x16x32_f16(false, af, false, bfrag, (short)0, acc2, false, false);
        }
        {
            v8h a0 = *(const v8h*)&Ash[48 + m16][kbA];
            v8h a1 = *(const v8h*)&Ash[48 + m16][kbA + 16];
            v16h af = __builtin_shufflevector(a0, a1,
                0, 1, 2, 3, 4, 5, 6, 7, 8, 9, 10, 11, 12, 13, 14, 15);
            acc3 = __builtin_amdgcn_wmma_f32_16x16x32_f16(false, af, false, bfrag, (short)0, acc3, false, false);
        }
        __syncthreads();
    }

    // epilogue: lanes 0-15 -> M=r, lanes 16-31 -> M=8+r; N = tileN + nrow
    const int gn = tileN + nrow;
    const int mb = (lane < 16) ? 0 : 8;
    if (gn < N) {
        const float bv = bias ? bias[gn] : 0.f;
        v8f* accs[4] = {&acc0, &acc1, &acc2, &acc3};
#pragma unroll
        for (int mt = 0; mt < 4; ++mt) {
            v8f a = *accs[mt];
#pragma unroll
            for (int r = 0; r < 8; ++r) {
                int gm = tileM + (mt << 4) + mb + r;
                if (gm < M) {
                    float v = alpha * a[r] + bv;
                    if (relu) v = fmaxf(v, 0.f);
                    C[(long)gm * ldc + gn] = v;
                }
            }
        }
    }
}

// ---------------------------------------------------------------------------
// Elementwise / small kernels
// ---------------------------------------------------------------------------
__global__ void imu_transpose(const float* __restrict__ imu, float* __restrict__ sensor,
                              int B, int L, int D)
{
    long idx = (long)blockIdx.x * blockDim.x + threadIdx.x;
    long tot = (long)B * D * L;
    if (idx >= tot) return;
    int l = idx % L; long r = idx / L;
    int c = r % D;   int b = r / D;
    sensor[idx] = imu[((long)b * L + l) * D + c];
}

__global__ void conv1d_k32(const float* __restrict__ x, const float* __restrict__ w,
                           float* __restrict__ y, int B, int L, int Kk, int coff, int Ctot)
{
    long idx = (long)blockIdx.x * blockDim.x + threadIdx.x;
    long tot = (long)B * 32 * L;
    if (idx >= tot) return;
    int l = idx % L; long r = idx / L;
    int co = r % 32; int b = r / 32;
    int half = (Kk - 1) / 2;
    float acc = 0.f;
    for (int ci = 0; ci < 32; ++ci) {
        const float* xr = x + ((long)b * 32 + ci) * L;
        const float* wr = w + ((long)co * 32 + ci) * Kk;
        for (int k = 0; k < Kk; ++k) {
            int ll = l + k - half;
            if (ll >= 0 && ll < L) acc += xr[ll] * wr[k];
        }
    }
    y[((long)b * Ctot + coff + co) * L + l] = acc;
}

__global__ void maxpool3(const float* __restrict__ x, float* __restrict__ y,
                         int B, int C, int L)
{
    long idx = (long)blockIdx.x * blockDim.x + threadIdx.x;
    long tot = (long)B * C * L;
    if (idx >= tot) return;
    int l = idx % L; long bc = idx / L;
    const float* xr = x + bc * L;
    float m = xr[l];
    if (l - 1 >= 0) m = fmaxf(m, xr[l - 1]);
    if (l + 1 < L)  m = fmaxf(m, xr[l + 1]);
    y[idx] = m;
}

__global__ void bn_act(float* __restrict__ y, const float* __restrict__ g,
                       const float* __restrict__ b, const float* __restrict__ res,
                       int B, int C, int L, int relu)
{
    long idx = (long)blockIdx.x * blockDim.x + threadIdx.x;
    long tot = (long)B * C * L;
    if (idx >= tot) return;
    int c = (idx / L) % C;
    float v = y[idx] * g[c] + b[c];
    if (res) v += res[idx];
    if (relu) v = fmaxf(v, 0.f);
    y[idx] = v;
}

__global__ void build_tokens(const float* __restrict__ xT, const float* __restrict__ pb,
                             const float* __restrict__ cls, const unsigned char* __restrict__ pm,
                             float* __restrict__ x, int B, int NT, int D, int L)
{
    long idx = (long)blockIdx.x * blockDim.x + threadIdx.x;
    long tot = (long)B * NT * D;
    if (idx >= tot) return;
    int d = idx % D; long r = idx / D;
    int t = r % NT;  int b = r / NT;
    if (t == 0) { x[idx] = cls[d]; return; }
    int l = t - 1;
    float m = pm[(long)b * L + l] ? 1.f : 0.f;
    x[idx] = (xT[((long)b * D + d) * L + l] + pb[d]) * m;
}

__global__ void wav_build(const float* __restrict__ w0, const float* __restrict__ w1,
                          const float* __restrict__ w2, float* __restrict__ kern, int D)
{
    int idx = blockIdx.x * blockDim.x + threadIdx.x;
    if (idx >= D * 19) return;
    int k = idx % 19, d = idx / 19;
    const float C = 0.8673250706f;  // 2/(3^0.5 * pi^0.25)
    const float* ws[3] = {w0, w1, w2};
    float acc = 0.f;
    for (int wv = 0; wv < 3; ++wv) {
        float s = ws[wv][d];        // scale plane
        float t = ws[wv][D + d];    // shift plane
        float xv = (float)(k - 9) - t;
        float u = xv / s;
        acc += C * (1.f - u * u) * __expf(-0.5f * u * u) * rsqrtf(fabsf(s));
    }
    kern[idx] = acc;
}

__global__ void tok2cmajor(const float* __restrict__ x, float* __restrict__ xc,
                           int B, int NT, int D, int L)
{
    long idx = (long)blockIdx.x * blockDim.x + threadIdx.x;
    long tot = (long)B * D * L;
    if (idx >= tot) return;
    int l = idx % L; long r = idx / L;
    int d = r % D;   int b = r / D;
    xc[idx] = x[((long)b * NT + 1 + l) * D + d];
}

__global__ void dwconv19(const float* __restrict__ xc, const float* __restrict__ kern,
                         float* __restrict__ pos, int B, int D, int L)
{
    long idx = (long)blockIdx.x * blockDim.x + threadIdx.x;
    long tot = (long)B * D * L;
    if (idx >= tot) return;
    int l = idx % L; long r = idx / L;
    int d = r % D;
    const float* xr = xc + (idx - l);
    const float* kr = kern + (long)d * 19;
    float acc = 0.f;
    for (int k = 0; k < 19; ++k) {
        int ll = l + k - 9;
        if (ll >= 0 && ll < L) acc += kr[k] * xr[ll];
    }
    pos[idx] = acc;
}

__global__ void add_enc(float* __restrict__ x, const float* __restrict__ posw,
                        const float* __restrict__ eb, int B, int NT, int D, int L)
{
    long idx = (long)blockIdx.x * blockDim.x + threadIdx.x;
    long tot = (long)B * L * D;
    if (idx >= tot) return;
    int d = idx % D; long r = idx / D;
    int l = r % L;   int b = r / L;
    x[((long)b * NT + 1 + l) * D + d] += posw[((long)b * D + d) * L + l] + eb[d];
}

__global__ void layernorm_rows(const float* __restrict__ x, float* __restrict__ y,
                               const float* __restrict__ g, const float* __restrict__ b,
                               int D)
{
    __shared__ float red[256];
    long row = blockIdx.x;
    const float* xr = x + row * D;
    float s = 0.f;
    for (int i = threadIdx.x; i < D; i += 256) s += xr[i];
    red[threadIdx.x] = s; __syncthreads();
    for (int o = 128; o > 0; o >>= 1) { if (threadIdx.x < o) red[threadIdx.x] += red[threadIdx.x + o]; __syncthreads(); }
    float mu = red[0] / D; __syncthreads();
    float v = 0.f;
    for (int i = threadIdx.x; i < D; i += 256) { float d0 = xr[i] - mu; v += d0 * d0; }
    red[threadIdx.x] = v; __syncthreads();
    for (int o = 128; o > 0; o >>= 1) { if (threadIdx.x < o) red[threadIdx.x] += red[threadIdx.x + o]; __syncthreads(); }
    float inv = rsqrtf(red[0] / D + 1e-5f);
    float* yr = y + row * D;
    for (int i = threadIdx.x; i < D; i += 256) yr[i] = (xr[i] - mu) * inv * g[i] + b[i];
}

__global__ void build_maskp(const unsigned char* __restrict__ pm, float* __restrict__ mp,
                            int B, int NP, int L)
{
    int idx = blockIdx.x * blockDim.x + threadIdx.x;
    if (idx >= B * NP) return;
    int i = idx % NP, b = idx / NP;
    float v;
    if (i < 255) v = 0.f;
    else if (i == 255) v = 1.f;
    else v = pm[(long)b * L + (i - 256)] ? 1.f : 0.f;
    mp[idx] = v;
}

__global__ void pad_tokens(const float* __restrict__ xn, float* __restrict__ xp,
                           int B, int NP, int NT, int D)
{
    long idx = (long)blockIdx.x * blockDim.x + threadIdx.x;
    long tot = (long)B * NP * D;
    if (idx >= tot) return;
    int d = idx % D; long r = idx / D;
    int i = r % NP;  int b = r / NP;
    xp[idx] = (i >= 255) ? xn[((long)b * NT + (i - 255)) * D + d] : 0.f;
}

__global__ void split_qkv(const float* __restrict__ qkv, const float* __restrict__ mp,
                          float* __restrict__ q, float* __restrict__ k, float* __restrict__ v,
                          int B, int H, int NP, int DH)
{
    long idx = (long)blockIdx.x * blockDim.x + threadIdx.x;
    long tot = (long)B * H * NP * DH;
    if (idx >= tot) return;
    int d = idx % DH; long r = idx / DH;
    int i = r % NP;   long r2 = r / NP;
    int h = r2 % H;   int b = r2 / H;
    long base = ((long)b * NP + i) * (3 * H * DH);
    float m = mp[(long)b * NP + i];
    q[idx] = qkv[base + h * DH + d] * m * 0.125f;               // * DH^-0.5
    k[idx] = qkv[base + H * DH + h * DH + d] * m;
    v[idx] = qkv[base + 2 * H * DH + h * DH + d] * m;
}

__global__ void msum_k(const float* __restrict__ mp, float* __restrict__ ms,
                       int B, int M, int NP)
{
    int idx = blockIdx.x * blockDim.x + threadIdx.x;
    if (idx >= B * M) return;
    int j = idx % M, b = idx / M;
    const float* r = mp + (long)b * NP + 3 * j;
    ms[idx] = r[0] + r[1] + r[2];
}

__global__ void landmark(const float* __restrict__ q, const float* __restrict__ ms,
                         float* __restrict__ ql, int BH, int M, int NP, int DH)
{
    long idx = (long)blockIdx.x * blockDim.x + threadIdx.x;
    long tot = (long)BH * M * DH;
    if (idx >= tot) return;
    int d = idx % DH; long r = idx / DH;
    int j = r % M;    int bh = r / M;
    int b = bh >> 3;
    const float* qb = q + ((long)bh * NP + 3 * j) * DH + d;
    float s = qb[0] + qb[DH] + qb[2 * DH];
    ql[idx] = s / (ms[(long)b * M + j] + 1e-8f);
}

__global__ void masked_softmax(float* __restrict__ S, const float* __restrict__ rmask,
                               const float* __restrict__ cmask, int R, int C,
                               long rms, long cms)
{
    __shared__ float red[256];
    int row = blockIdx.x;
    int bh  = blockIdx.y;
    int b   = bh >> 3;
    float* Sr = S + ((long)bh * R + row) * C;
    const float BIGNEG = -3.402823466e+38f;
    bool rm = rmask[(long)b * rms + row] > 0.f;
    float mx = BIGNEG;
    for (int j = threadIdx.x; j < C; j += 256) {
        float v = (rm && cmask[(long)b * cms + j] > 0.f) ? Sr[j] : BIGNEG;
        Sr[j] = v;
        mx = fmaxf(mx, v);
    }
    red[threadIdx.x] = mx; __syncthreads();
    for (int o = 128; o > 0; o >>= 1) { if (threadIdx.x < o) red[threadIdx.x] = fmaxf(red[threadIdx.x], red[threadIdx.x + o]); __syncthreads(); }
    mx = red[0]; __syncthreads();
    float sum = 0.f;
    for (int j = threadIdx.x; j < C; j += 256) { float e = __expf(Sr[j] - mx); Sr[j] = e; sum += e; }
    red[threadIdx.x] = sum; __syncthreads();
    for (int o = 128; o > 0; o >>= 1) { if (threadIdx.x < o) red[threadIdx.x] += red[threadIdx.x + o]; __syncthreads(); }
    float inv = 1.f / red[0];
    for (int j = threadIdx.x; j < C; j += 256) Sr[j] *= inv;
}

__global__ void zero_scal(float* s) { if (threadIdx.x < 2) s[threadIdx.x] = 0.f; }

__global__ void abssum_max(const float* __restrict__ X, float* __restrict__ scal,
                           int BH, int Nn, int mode)
{
    int idx = blockIdx.x * blockDim.x + threadIdx.x;
    if (idx >= BH * Nn) return;
    int r = idx % Nn, bh = idx / Nn;
    const float* base = X + (long)bh * Nn * Nn;
    float s = 0.f;
    if (mode == 0) { for (int j = 0; j < Nn; ++j) s += fabsf(base[(long)r * Nn + j]); }
    else           { for (int i = 0; i < Nn; ++i) s += fabsf(base[(long)i * Nn + r]); }
    atomicMax((int*)(scal + mode), __float_as_int(s));
}

__global__ void pinv_init(const float* __restrict__ X, const float* __restrict__ scal,
                          float* __restrict__ z, int BH, int Nn)
{
    long idx = (long)blockIdx.x * blockDim.x + threadIdx.x;
    long tot = (long)BH * Nn * Nn;
    if (idx >= tot) return;
    int j = idx % Nn; long r = idx / Nn;
    int i = r % Nn;   int bh = r / Nn;
    float inv = 1.f / (scal[0] * scal[1]);
    z[idx] = X[((long)bh * Nn + j) * Nn + i] * inv;
}

__global__ void aI_minus(const float* __restrict__ X, float* __restrict__ E,
                         float alpha, int BH, int Nn)
{
    long idx = (long)blockIdx.x * blockDim.x + threadIdx.x;
    long tot = (long)BH * Nn * Nn;
    if (idx >= tot) return;
    int j = idx % Nn;
    int i = (idx / Nn) % Nn;
    E[idx] = ((i == j) ? alpha : 0.f) - X[idx];
}

__global__ void resconv_add(float* __restrict__ O, const float* __restrict__ v,
                            const float* __restrict__ rw, int BH, int NP, int DH)
{
    long idx = (long)blockIdx.x * blockDim.x + threadIdx.x;
    long tot = (long)BH * NP * DH;
    if (idx >= tot) return;
    int d = idx % DH; long r = idx / DH;
    int i = r % NP;   int bh = r / NP;
    int h = bh & 7;
    float acc = 0.f;
    for (int t = 0; t < 33; ++t) {
        int ii = i + t - 16;
        if (ii >= 0 && ii < NP) acc += rw[h * 33 + t] * v[((long)bh * NP + ii) * DH + d];
    }
    O[idx] += acc;
}

__global__ void merge_heads(const float* __restrict__ O, float* __restrict__ mrg,
                            int B, int H, int NP, int DH)
{
    long idx = (long)blockIdx.x * blockDim.x + threadIdx.x;
    long tot = (long)B * NP * H * DH;
    if (idx >= tot) return;
    int hd = idx % (H * DH); long r = idx / (H * DH);
    int i = r % NP;          int b = r / NP;
    int h = hd / DH, d = hd % DH;
    mrg[idx] = O[(((long)b * H + h) * NP + i) * DH + d];
}

__global__ void slice_add(float* __restrict__ x, const float* __restrict__ attn,
                          int B, int NT, int NP, int D)
{
    long idx = (long)blockIdx.x * blockDim.x + threadIdx.x;
    long tot = (long)B * NT * D;
    if (idx >= tot) return;
    int d = idx % D; long r = idx / D;
    int t = r % NT;  int b = r / NT;
    x[idx] += attn[((long)b * NP + 255 + t) * D + d];
}

// ---------------------------------------------------------------------------
// Host orchestration
// ---------------------------------------------------------------------------
static void gemm(hipStream_t st, const float* A, const float* B, float* C,
                 int M, int N, int K, long lda, long ldb, long ldc,
                 long sA, long sB, long sC, int batch, int transB,
                 const float* bias, float alpha, int relu)
{
    dim3 grid(divup(N, 64), divup(M, 64), batch);
    wmma_gemm<<<grid, 128, 0, st>>>(A, B, C, M, N, K, lda, ldb, ldc,
                                    sA, sB, sC, transB, bias, alpha, relu);
}

extern "C" void kernel_launch(void* const* d_in, const int* in_sizes, int n_in,
                              void* d_out, int out_size, void* d_ws, size_t ws_size,
                              hipStream_t stream)
{
    const int Bn = 16, L = 512, DIN = 7, NF = 128, D = 512;
    const int NT = 513, NP = 768, Mland = 256, H = 8, DH = 64, BH = 128;

    const float* imu = (const float*)d_in[0];
    const unsigned char* pm = (const unsigned char*)d_in[1];
    const float* proj_w = (const float*)d_in[50];
    const float* proj_b = (const float*)d_in[51];
    const float* cls    = (const float*)d_in[52];

    float* wsf = (float*)d_ws;
    size_t off = 0;
    auto alloc = [&](size_t n) { float* p = wsf + off; off += (n + 31) & ~(size_t)31; return p; };

    float* sensor = alloc((size_t)Bn * DIN * L);
    float* f0     = alloc((size_t)Bn * NF * L);
    float* f1     = alloc((size_t)Bn * NF * L);
    float* btl    = alloc((size_t)Bn * 32 * L);
    float* poolb  = alloc((size_t)Bn * NF * L);
    float* resb   = alloc((size_t)Bn * NF * L);
    float* prevb  = alloc((size_t)Bn * NF * L);
    float* kernb  = alloc((size_t)D * 19);
    float* xT     = alloc((size_t)Bn * D * L);
    float* x      = alloc((size_t)Bn * NT * D);
    float* xc     = alloc((size_t)Bn * D * L);
    float* pos    = alloc((size_t)Bn * D * L);
    float* posw   = alloc((size_t)Bn * D * L);
    float* xn     = alloc((size_t)Bn * NT * D);
    float* maskp  = alloc((size_t)Bn * NP);
    float* msum   = alloc((size_t)Bn * Mland);
    float* xp     = alloc((size_t)Bn * NP * D);
    float* qkv    = alloc((size_t)Bn * NP * 3 * D);
    float* q      = alloc((size_t)BH * NP * DH);
    float* k      = alloc((size_t)BH * NP * DH);
    float* v      = alloc((size_t)BH * NP * DH);
    float* ql     = alloc((size_t)BH * Mland * DH);
    float* kl     = alloc((size_t)BH * Mland * DH);
    float* s1     = alloc((size_t)BH * NP * Mland);
    float* s2     = alloc((size_t)BH * Mland * Mland);
    float* s3     = alloc((size_t)BH * Mland * NP);
    float* z_a    = alloc((size_t)BH * Mland * Mland);
    float* z_b    = alloc((size_t)BH * Mland * Mland);
    float* Pb     = alloc((size_t)BH * Mland * Mland);
    float* Eb     = alloc((size_t)BH * Mland * Mland);
    float* Fb     = alloc((size_t)BH * Mland * Mland);
    float* Gb     = alloc((size_t)BH * Mland * Mland);
    float* av     = alloc((size_t)BH * Mland * DH);
    float* zav    = alloc((size_t)BH * Mland * DH);
    float* Ob     = alloc((size_t)BH * NP * DH);
    float* mrg    = alloc((size_t)Bn * NP * D);
    float* attnb  = alloc((size_t)Bn * NP * D);
    float* h1     = alloc((size_t)Bn * D);
    float* scal   = alloc(32);
    (void)ws_size; (void)in_sizes; (void)n_in;

    auto g1 = [](long n) { return dim3(divup(n, 256)); };

    // ---------------- sensor + inception stack ----------------
    imu_transpose<<<g1((long)Bn * DIN * L), 256, 0, stream>>>(imu, sensor, Bn, L, DIN);

    const float* xsrc = sensor; int cin = DIN;
    const float* prev = sensor; int cinp = DIN;
    for (int i = 0; i < 6; ++i) {
        int pb = 2 + i * 7;
        const float* wb  = (const float*)d_in[pb + 0];
        const float* w39 = (const float*)d_in[pb + 1];
        const float* w19 = (const float*)d_in[pb + 2];
        const float* w9  = (const float*)d_in[pb + 3];
        const float* wp  = (const float*)d_in[pb + 4];
        const float* bg  = (const float*)d_in[pb + 5];
        const float* bb  = (const float*)d_in[pb + 6];
        float* y = (i & 1) ? f1 : f0;

        // bottleneck 1x1 as batched WMMA GEMM: (32 x cin) @ (cin x L)
        gemm(stream, wb, xsrc, btl, 32, L, cin, cin, L, L,
             0, (long)cin * L, 32L * L, Bn, 0, nullptr, 1.f, 0);
        // wide convs into channel slices 0/32/64
        conv1d_k32<<<g1((long)Bn * 32 * L), 256, 0, stream>>>(btl, w39, y, Bn, L, 39, 0, NF);
        conv1d_k32<<<g1((long)Bn * 32 * L), 256, 0, stream>>>(btl, w19, y, Bn, L, 19, 32, NF);
        conv1d_k32<<<g1((long)Bn * 32 * L), 256, 0, stream>>>(btl, w9,  y, Bn, L,  9, 64, NF);
        // maxpool branch (1x1 conv as GEMM) into channels 96..127
        maxpool3<<<g1((long)Bn * cin * L), 256, 0, stream>>>(xsrc, poolb, Bn, cin, L);
        gemm(stream, wp, poolb, y + 96L * L, 32, L, cin, cin, L, L,
             0, (long)cin * L, (long)NF * L, Bn, 0, nullptr, 1.f, 0);

        if (i % 3 == 2) {
            int ri = i / 3;
            const float* rw = (const float*)d_in[44 + ri * 3];
            const float* rg = (const float*)d_in[45 + ri * 3];
            const float* rb = (const float*)d_in[46 + ri * 3];
            gemm(stream, rw, prev, resb, NF, L, cinp, cinp, L, L,
                 0, (long)cinp * L, (long)NF * L, Bn, 0, nullptr, 1.f, 0);
            bn_act<<<g1((long)Bn * NF * L), 256, 0, stream>>>(resb, rg, rb, nullptr, Bn, NF, L, 0);
            bn_act<<<g1((long)Bn * NF * L), 256, 0, stream>>>(y, bg, bb, resb, Bn, NF, L, 1);
            if (i == 2) {
                hipMemcpyAsync(prevb, y, (size_t)Bn * NF * L * sizeof(float),
                               hipMemcpyDeviceToDevice, stream);
                prev = prevb; cinp = NF;
            }
        } else {
            bn_act<<<g1((long)Bn * NF * L), 256, 0, stream>>>(y, bg, bb, nullptr, Bn, NF, L, 1);
        }
        xsrc = y; cin = NF;
    }

    // ---------------- projection + cls token ----------------
    gemm(stream, proj_w, xsrc, xT, D, L, NF, NF, L, L,
         0, (long)NF * L, (long)D * L, Bn, 0, nullptr, 1.f, 0);
    build_tokens<<<g1((long)Bn * NT * D), 256, 0, stream>>>(xT, proj_b, cls, pm, x, Bn, NT, D, L);

    // ---------------- 2x (wavelet encoding + Nystrom translayer) ----------------
    for (int ly = 0; ly < 2; ++ly) {
        // wavelet positional encoding
        const float* wv0 = (const float*)d_in[53 + ly * 3 + 0];
        const float* wv1 = (const float*)d_in[53 + ly * 3 + 1];
        const float* wv2 = (const float*)d_in[53 + ly * 3 + 2];
        const float* encw = (const float*)d_in[59 + ly * 2];
        const float* encb = (const float*)d_in[60 + ly * 2];
        wav_build<<<g1((long)D * 19), 256, 0, stream>>>(wv0, wv1, wv2, kernb, D);
        tok2cmajor<<<g1((long)Bn * D * L), 256, 0, stream>>>(x, xc, Bn, NT, D, L);
        dwconv19<<<g1((long)Bn * D * L), 256, 0, stream>>>(xc, kernb, pos, Bn, D, L);
        gemm(stream, encw, pos, posw, D, L, D, D, L, L,
             0, (long)D * L, (long)D * L, Bn, 0, nullptr, 1.f, 0);
        add_enc<<<g1((long)Bn * L * D), 256, 0, stream>>>(x, posw, encb, Bn, NT, D, L);

        // translayer
        const float* ln_g = (const float*)d_in[63 + ly * 6 + 0];
        const float* ln_b = (const float*)d_in[63 + ly * 6 + 1];
        const float* wqkv = (const float*)d_in[63 + ly * 6 + 2];
        const float* wo   = (const float*)d_in[63 + ly * 6 + 3];
        const float* bo   = (const float*)d_in[63 + ly * 6 + 4];
        const float* rcw  = (const float*)d_in[63 + ly * 6 + 5];

        layernorm_rows<<<dim3((unsigned)(Bn * NT)), 256, 0, stream>>>(x, xn, ln_g, ln_b, D);
        build_maskp<<<g1((long)Bn * NP), 256, 0, stream>>>(pm, maskp, Bn, NP, L);
        pad_tokens<<<g1((long)Bn * NP * D), 256, 0, stream>>>(xn, xp, Bn, NP, NT, D);

        // QKV: (B*768 x 512) @ wqkv^T (1536 x 512)
        gemm(stream, xp, wqkv, qkv, Bn * NP, 3 * D, D, D, D, 3 * D,
             0, 0, 0, 1, 1, nullptr, 1.f, 0);
        split_qkv<<<g1((long)BH * NP * DH), 256, 0, stream>>>(qkv, maskp, q, k, v, Bn, H, NP, DH);

        msum_k<<<g1((long)Bn * Mland), 256, 0, stream>>>(maskp, msum, Bn, Mland, NP);
        landmark<<<g1((long)BH * Mland * DH), 256, 0, stream>>>(q, msum, ql, BH, Mland, NP, DH);
        landmark<<<g1((long)BH * Mland * DH), 256, 0, stream>>>(k, msum, kl, BH, Mland, NP, DH);

        // s1 = q @ kl^T ; s2 = ql @ kl^T ; s3 = ql @ k^T   (batched over BH)
        gemm(stream, q, kl, s1, NP, Mland, DH, DH, DH, Mland,
             (long)NP * DH, (long)Mland * DH, (long)NP * Mland, BH, 1, nullptr, 1.f, 0);
        gemm(stream, ql, kl, s2, Mland, Mland, DH, DH, DH, Mland,
             (long)Mland * DH, (long)Mland * DH, (long)Mland * Mland, BH, 1, nullptr, 1.f, 0);
        gemm(stream, ql, k, s3, Mland, NP, DH, DH, DH, NP,
             (long)Mland * DH, (long)NP * DH, (long)Mland * NP, BH, 1, nullptr, 1.f, 0);

        masked_softmax<<<dim3((unsigned)NP, (unsigned)BH), 256, 0, stream>>>(s1, maskp, msum, NP, Mland, NP, Mland);
        masked_softmax<<<dim3((unsigned)Mland, (unsigned)BH), 256, 0, stream>>>(s2, msum, msum, Mland, Mland, Mland, Mland);
        masked_softmax<<<dim3((unsigned)Mland, (unsigned)BH), 256, 0, stream>>>(s3, msum, maskp, Mland, NP, Mland, NP);

        // av = a3 @ v  (before pinv so s3 stays read-only)
        gemm(stream, s3, v, av, Mland, DH, NP, NP, DH, DH,
             (long)Mland * NP, (long)NP * DH, (long)Mland * DH, BH, 0, nullptr, 1.f, 0);

        // Moore-Penrose pinv of a2 (Newton-Schulz, 6 iters, all WMMA GEMMs)
        zero_scal<<<1, 32, 0, stream>>>(scal);
        abssum_max<<<g1((long)BH * Mland), 256, 0, stream>>>(s2, scal, BH, Mland, 0);
        abssum_max<<<g1((long)BH * Mland), 256, 0, stream>>>(s2, scal, BH, Mland, 1);
        pinv_init<<<g1((long)BH * Mland * Mland), 256, 0, stream>>>(s2, scal, z_a, BH, Mland);

        float* zc = z_a; float* zo = z_b;
        const long sM2 = (long)Mland * Mland;
        for (int it = 0; it < 6; ++it) {
            gemm(stream, s2, zc, Pb, Mland, Mland, Mland, Mland, Mland, Mland,
                 sM2, sM2, sM2, BH, 0, nullptr, 1.f, 0);                    // P = x@z
            aI_minus<<<g1((long)BH * sM2), 256, 0, stream>>>(Pb, Eb, 7.f, BH, Mland);
            gemm(stream, Pb, Eb, Fb, Mland, Mland, Mland, Mland, Mland, Mland,
                 sM2, sM2, sM2, BH, 0, nullptr, 1.f, 0);                    // F = P@(7I-P)
            aI_minus<<<g1((long)BH * sM2), 256, 0, stream>>>(Fb, Eb, 15.f, BH, Mland);
            gemm(stream, Pb, Eb, Gb, Mland, Mland, Mland, Mland, Mland, Mland,
                 sM2, sM2, sM2, BH, 0, nullptr, 1.f, 0);                    // G = P@(15I-F)
            aI_minus<<<g1((long)BH * sM2), 256, 0, stream>>>(Gb, Fb, 13.f, BH, Mland);
            gemm(stream, zc, Fb, zo, Mland, Mland, Mland, Mland, Mland, Mland,
                 sM2, sM2, sM2, BH, 0, nullptr, 0.25f, 0);                  // z' = 0.25*z@(13I-G)
            float* t = zc; zc = zo; zo = t;
        }

        // out = a1 @ (z @ av) + depthwise residual conv on v
        gemm(stream, zc, av, zav, Mland, DH, Mland, Mland, DH, DH,
             sM2, (long)Mland * DH, (long)Mland * DH, BH, 0, nullptr, 1.f, 0);
        gemm(stream, s1, zav, Ob, NP, DH, Mland, Mland, DH, DH,
             (long)NP * Mland, (long)Mland * DH, (long)NP * DH, BH, 0, nullptr, 1.f, 0);
        resconv_add<<<g1((long)BH * NP * DH), 256, 0, stream>>>(Ob, v, rcw, BH, NP, DH);
        merge_heads<<<g1((long)Bn * NP * D), 256, 0, stream>>>(Ob, mrg, Bn, H, NP, DH);
        gemm(stream, mrg, wo, attnb, Bn * NP, D, D, D, D, D,
             0, 0, 0, 1, 1, bo, 1.f, 0);
        slice_add<<<g1((long)Bn * NT * D), 256, 0, stream>>>(x, attnb, Bn, NT, NP, D);
    }

    // ---------------- classification heads (cls token) ----------------
    float* out = (float*)d_out;
    const float* f2w1 = (const float*)d_in[75];
    const float* f2b1 = (const float*)d_in[76];
    const float* f2w2 = (const float*)d_in[77];
    const float* f2b2 = (const float*)d_in[78];
    const float* faw1 = (const float*)d_in[79];
    const float* fab1 = (const float*)d_in[80];
    const float* faw2 = (const float*)d_in[81];
    const float* fab2 = (const float*)d_in[82];

    // cls rows of x have leading dim NT*D
    gemm(stream, x, f2w1, h1, Bn, D, D, (long)NT * D, D, D, 0, 0, 0, 1, 1, f2b1, 1.f, 1);
    gemm(stream, h1, f2w2, out, Bn, 18, D, D, D, 18, 0, 0, 0, 1, 1, f2b2, 1.f, 0);
    gemm(stream, x, faw1, h1, Bn, D, D, (long)NT * D, D, D, 0, 0, 0, 1, 1, fab1, 1.f, 1);
    gemm(stream, h1, faw2, out + (long)Bn * 18, Bn, 2, D, D, D, 2, 0, 0, 0, 1, 1, fab2, 1.f, 0);
    (void)out_size;
}
